// GATClassifier_41497974014276
// MI455X (gfx1250) — matrix operand order repaired and checked by reading, not compiled, not run on hardware
//
#include <hip/hip_runtime.h>
#include <math.h>

// ---------------------------------------------------------------------------
// GAT classifier for MI455X (gfx1250, wave32).
//   layer1: feat1 = x@W1 (WMMA f16->f32), 4-head edge softmax, agg, bias+relu
//   layer2: feat2 = h1@W2 (WMMA),        1-head edge softmax, agg
//   out    = (agg2 + b2) @ fc_W + fc_b - thres
// ---------------------------------------------------------------------------

typedef __attribute__((ext_vector_type(16))) _Float16 v16h;
typedef __attribute__((ext_vector_type(8)))  _Float16 v8h;
typedef __attribute__((ext_vector_type(4)))  _Float16 v4h;
typedef __attribute__((ext_vector_type(8)))  float    v8f;
typedef __attribute__((ext_vector_type(4)))  float    v4f;

union V16 { v16h v; v8h h[2]; };

#define DIM 128   // IN_DIM == HID == HEADS*OUT_F == 128 everywhere

// ---------------------------------------------------------------------------
// C[N x 128] = A[N x 128] * W[128 x 128], f32 in/out, f16 WMMA with f32 acc.
// Block: 128 threads = 4 waves. Each block does a 64-row slab, each wave a
// 16-row strip across all 8 column tiles. K = 128 = 4 steps of 32.
// W is staged into LDS pre-swizzled into WMMA B-fragment order so each lane's
// fragment is 32 contiguous bytes (2x ds_load_b128 per fragment).
// ---------------------------------------------------------------------------
__global__ __launch_bounds__(128)
void gemm128_wmma(const float* __restrict__ A, const float* __restrict__ W,
                  float* __restrict__ C, int nrows) {
  // B-fragment layout: half index = ((frag*32 + lane)*16 + j),
  //   frag = nt*4 + kt;  lane = grp*16 + m;  j = 0..15
  //   holds W[kt*32 + grp*16 + j, nt*16 + m]
  __shared__ _Float16 lds_bw[8 * 4 * 32 * 16];  // 32 KB
  __shared__ _Float16 lds_a[64 * DIM];          // 16 KB, row-major
  const int tid  = threadIdx.x;
  const int lane = tid & 31;
  const int wave = tid >> 5;                    // 0..3

  // ---- stage W (coalesced float4 reads, swizzled f16 scatter into LDS) ----
  for (int i = tid; i < DIM * DIM / 4; i += 128) {
    v4f w4 = ((const v4f*)W)[i];
    int flat = i * 4;                 // = k*128 + n
    int k = flat >> 7, n = flat & (DIM - 1);
    int nt = n >> 4, kt = k >> 5, grp = (k >> 4) & 1, j = k & 15;
    int base = ((nt * 4 + kt) * 32 + grp * 16 + (n & 15)) * 16 + j;
#pragma unroll
    for (int q = 0; q < 4; ++q)       // n..n+3 -> consecutive lanes, same j
      lds_bw[base + q * 16] = (_Float16)w4[q];
  }

  // ---- stage 64-row activation slab (float4 reads, v4h stores) ----
  const int row0 = blockIdx.x * 64;
  for (int i = tid; i < 64 * DIM / 4; i += 128) {
    int flat = i * 4;
    int r = flat >> 7, c = flat & (DIM - 1);
    int gr = row0 + r;
    v4f x4;
    if (gr < nrows) x4 = ((const v4f*)A)[((size_t)gr * DIM + c) >> 2];
    else            x4 = (v4f){0.f, 0.f, 0.f, 0.f};
    v4h x4h;
#pragma unroll
    for (int q = 0; q < 4; ++q) x4h[q] = (_Float16)x4[q];
    *(v4h*)&lds_a[r * DIM + c] = x4h;
  }
  __syncthreads();

  const int m   = lane & 15;          // A row / C,D column within the 16x16 tile
  const int grp = lane >> 4;          // lane group 0/1
  const int koA = grp * 16;           // A: contiguous K per lane group (matches B)
  const int arow = wave * 16 + m;

  // preload all 4 A K-fragments (reused by all 8 column tiles)
  V16 afrag[4];
#pragma unroll
  for (int kt = 0; kt < 4; ++kt) {
    const int kb = kt * 32;
    afrag[kt].h[0] = *(const v8h*)&lds_a[arow * DIM + kb + koA];       // K+0..7
    afrag[kt].h[1] = *(const v8h*)&lds_a[arow * DIM + kb + 8 + koA];  // K+8..15
  }

  const int rbase = row0 + wave * 16 + grp * 8;
  const bool full = (row0 + 64 <= nrows);   // uniform: slab fully in bounds
#pragma unroll
  for (int nt = 0; nt < 8; ++nt) {    // 8 column tiles of 16
    v8f acc = {};
#pragma unroll
    for (int kt = 0; kt < 4; ++kt) {  // K steps of 32
      V16 b;
      const v8h* bp = (const v8h*)&lds_bw[((nt * 4 + kt) * 32 + lane) * 16];
      b.h[0] = bp[0];
      b.h[1] = bp[1];
      acc = __builtin_amdgcn_wmma_f32_16x16x32_f16(
          /*neg_a=*/false, afrag[kt].v, /*neg_b=*/false, b.v,
          /*c_mod=*/(short)0, acc, /*reuse_a=*/false, /*reuse_b=*/false);
    }
    // D layout: VGPR i -> row (i + grp*8), col = lane&15
    float* cp = C + (size_t)rbase * DIM + nt * 16 + m;
    if (full) {                       // fast path: single scalar branch
#pragma unroll
      for (int i = 0; i < 8; ++i) cp[(size_t)i * DIM] = acc[i];
    } else {
#pragma unroll
      for (int i = 0; i < 8; ++i)
        if (rbase + i < nrows) cp[(size_t)i * DIM] = acc[i];
    }
  }
}

// ---------------------------------------------------------------------------
// el[n,h] = sum_f feat[n,h,f] * al[h,f];  er likewise.  One thread per (n,h).
// ---------------------------------------------------------------------------
__global__ void attn_scores(const float* __restrict__ feat,
                            const float* __restrict__ al,
                            const float* __restrict__ ar,
                            float* __restrict__ el, float* __restrict__ er,
                            int nh, int H, int F) {
  int t = blockIdx.x * blockDim.x + threadIdx.x;
  if (t >= nh) return;
  int h = t % H;
  const float* fp = feat + (size_t)t * F;
  const float* lp = al + h * F;
  const float* rp = ar + h * F;
  float sl = 0.f, sr = 0.f;
  for (int f = 0; f < F; ++f) { float v = fp[f]; sl += v * lp[f]; sr += v * rp[f]; }
  el[t] = sl; er[t] = sr;
}

__global__ void fillk(float* __restrict__ p, float v, unsigned n) {
  unsigned i = blockIdx.x * blockDim.x + threadIdx.x;
  if (i < n) p[i] = v;
}

// float atomic max via ordered int/uint bits (init must be -inf)
__device__ inline void atomicMaxF(float* addr, float v) {
  if (v >= 0.f) atomicMax((int*)addr, __float_as_int(v));
  else          atomicMin((unsigned int*)addr, (unsigned int)__float_as_int(v));
}

__device__ inline float lrelu(float v) { return v > 0.f ? v : 0.2f * v; }

// pass 1: segment max of leaky-relu edge scores over incoming edges per dst
__global__ void edge_max(const int* __restrict__ src, const int* __restrict__ dst,
                         const float* __restrict__ el, const float* __restrict__ er,
                         float* __restrict__ emax, int EH, int H) {
  int t = blockIdx.x * blockDim.x + threadIdx.x;
  if (t >= EH) return;
  int e = t / H, h = t - e * H;
  int s = src[e], d = dst[e];
  atomicMaxF(&emax[d * H + h], lrelu(el[s * H + h] + er[d * H + h]));
}

// pass 2: ex = exp(e - emax[dst]); denom[dst] += ex; store ex per edge
__global__ void edge_exp(const int* __restrict__ src, const int* __restrict__ dst,
                         const float* __restrict__ el, const float* __restrict__ er,
                         const float* __restrict__ emax, float* __restrict__ denom,
                         float* __restrict__ exbuf, int EH, int H) {
  int t = blockIdx.x * blockDim.x + threadIdx.x;
  if (t >= EH) return;
  int e = t / H, h = t - e * H;
  int s = src[e], d = dst[e];
  float v  = lrelu(el[s * H + h] + er[d * H + h]);
  float ex = __expf(v - emax[d * H + h]);
  exbuf[t] = ex;
  atomicAdd(&denom[d * H + h], ex);
}

// pass 2b: alpha = ex / denom[dst]  (in place on exbuf)
__global__ void edge_alpha(const int* __restrict__ dst,
                           const float* __restrict__ denom,
                           float* __restrict__ exbuf, int EH, int H) {
  int t = blockIdx.x * blockDim.x + threadIdx.x;
  if (t >= EH) return;
  int e = t / H, h = t - e * H;
  exbuf[t] = exbuf[t] / denom[dst[e] * H + h];
}

// pass 3: agg[dst, h, f] += feat[src, h, f] * alpha[e,h]; float4 per thread
__global__ void edge_agg(const int* __restrict__ src, const int* __restrict__ dst,
                         const float* __restrict__ alpha,
                         const float* __restrict__ feat, float* __restrict__ agg,
                         long total4, int H, int F) {
  long t = (long)blockIdx.x * blockDim.x + threadIdx.x;
  if (t >= total4) return;
  const int HFq = H * F / 4;
  long e  = t / HFq;
  int  c4 = (int)(t - e * HFq);
  int  c  = c4 * 4;
  int  h  = c / F;
  int  s = src[e], d = dst[e];
  float a = alpha[e * H + h];
  v4f f4 = ((const v4f*)feat)[((size_t)s * (H * F) + c) >> 2];
  float* out = &agg[(size_t)d * (H * F) + c];
#pragma unroll
  for (int q = 0; q < 4; ++q) atomicAdd(out + q, f4[q] * a);
}

// h = relu(agg + bias[H*F]) in place
__global__ void bias_relu(float* __restrict__ p, const float* __restrict__ bias,
                          unsigned n) {
  unsigned i = blockIdx.x * blockDim.x + threadIdx.x;
  if (i >= n) return;
  float v = p[i] + bias[i & (DIM - 1)];
  p[i] = v > 0.f ? v : 0.f;
}

// out[n] = sum_j (agg2[n,j] + b2[j]) * fcW[j] + fcb - thres.  Wave per node.
__global__ __launch_bounds__(128)
void finalize2(const float* __restrict__ agg, const float* __restrict__ b2,
               const float* __restrict__ fcW, const float* __restrict__ fcb,
               const float* __restrict__ thres, float* __restrict__ out, int n) {
  int node = blockIdx.x * (blockDim.x >> 5) + (threadIdx.x >> 5);
  int lane = threadIdx.x & 31;
  if (node >= n) return;
  float s = 0.f;
  for (int j = lane; j < DIM; j += 32)
    s += (agg[(size_t)node * DIM + j] + b2[j]) * fcW[j];
#pragma unroll
  for (int off = 16; off; off >>= 1) s += __shfl_xor(s, off, 32);
  if (lane == 0) out[node] = s + fcb[0] - thres[0];
}

// ---------------------------------------------------------------------------
extern "C" void kernel_launch(void* const* d_in, const int* in_sizes, int n_in,
                              void* d_out, int out_size, void* d_ws, size_t ws_size,
                              hipStream_t stream) {
  const float* features = (const float*)d_in[0];
  const float* W1  = (const float*)d_in[1];
  const float* al1 = (const float*)d_in[2];
  const float* ar1 = (const float*)d_in[3];
  const float* b1  = (const float*)d_in[4];
  const float* W2  = (const float*)d_in[5];
  const float* al2 = (const float*)d_in[6];
  const float* ar2 = (const float*)d_in[7];
  const float* b2  = (const float*)d_in[8];
  const float* fcW = (const float*)d_in[9];
  const float* fcb = (const float*)d_in[10];
  const float* thr = (const float*)d_in[11];
  const int*   src = (const int*)d_in[12];
  const int*   dst = (const int*)d_in[13];

  const int N = in_sizes[0] / DIM;     // 100000
  const int E = in_sizes[12];          // 1600000
  const int H1 = 4, F1 = 32, H2 = 1, F2 = 128;

  // workspace carve-up (floats)
  float* ws = (float*)d_ws;
  size_t o = 0;
  float* feat1  = ws + o; o += (size_t)N * DIM;
  float* el1    = ws + o; o += (size_t)N * H1;
  float* er1    = ws + o; o += (size_t)N * H1;
  float* emax1  = ws + o; o += (size_t)N * H1;
  float* denom1 = ws + o; o += (size_t)N * H1;
  float* ex1    = ws + o; o += (size_t)E * H1;
  float* agg1   = ws + o; o += (size_t)N * DIM;   // becomes h1 after bias_relu
  float* feat2  = ws + o; o += (size_t)N * DIM;
  float* el2    = ws + o; o += (size_t)N * H2;
  float* er2    = ws + o; o += (size_t)N * H2;
  float* emax2  = ws + o; o += (size_t)N * H2;
  float* denom2 = ws + o; o += (size_t)N * H2;
  float* ex2    = ws + o; o += (size_t)E * H2;
  float* agg2   = ws + o; o += (size_t)N * DIM;
  (void)ws_size; (void)n_in;

  const int T = 256;
  auto g = [](long n, int t) { return (unsigned)((n + t - 1) / t); };
  const unsigned gGemm = g(N, 64);

  // ---- layer 1 ----
  gemm128_wmma<<<gGemm, 128, 0, stream>>>(features, W1, feat1, N);
  attn_scores<<<g((long)N * H1, T), T, 0, stream>>>(feat1, al1, ar1, el1, er1, N * H1, H1, F1);
  fillk<<<g((long)N * H1, T), T, 0, stream>>>(emax1, -INFINITY, N * H1);
  fillk<<<g((long)N * H1, T), T, 0, stream>>>(denom1, 0.f, N * H1);
  fillk<<<g((long)N * DIM, T), T, 0, stream>>>(agg1, 0.f, (unsigned)N * DIM);
  edge_max<<<g((long)E * H1, T), T, 0, stream>>>(src, dst, el1, er1, emax1, E * H1, H1);
  edge_exp<<<g((long)E * H1, T), T, 0, stream>>>(src, dst, el1, er1, emax1, denom1, ex1, E * H1, H1);
  edge_alpha<<<g((long)E * H1, T), T, 0, stream>>>(dst, denom1, ex1, E * H1, H1);
  edge_agg<<<g((long)E * DIM / 4, T), T, 0, stream>>>(src, dst, ex1, feat1, agg1,
                                                     (long)E * DIM / 4, H1, F1);
  bias_relu<<<g((long)N * DIM, T), T, 0, stream>>>(agg1, b1, (unsigned)N * DIM);

  // ---- layer 2 ----
  gemm128_wmma<<<gGemm, 128, 0, stream>>>(agg1, W2, feat2, N);
  attn_scores<<<g((long)N * H2, T), T, 0, stream>>>(feat2, al2, ar2, el2, er2, N * H2, H2, F2);
  fillk<<<g((long)N * H2, T), T, 0, stream>>>(emax2, -INFINITY, N * H2);
  fillk<<<g((long)N * H2, T), T, 0, stream>>>(denom2, 0.f, N * H2);
  fillk<<<g((long)N * DIM, T), T, 0, stream>>>(agg2, 0.f, (unsigned)N * DIM);
  edge_max<<<g((long)E * H2, T), T, 0, stream>>>(src, dst, el2, er2, emax2, E * H2, H2);
  edge_exp<<<g((long)E * H2, T), T, 0, stream>>>(src, dst, el2, er2, emax2, denom2, ex2, E * H2, H2);
  edge_alpha<<<g((long)E * H2, T), T, 0, stream>>>(dst, denom2, ex2, E * H2, H2);
  edge_agg<<<g((long)E * DIM / 4, T), T, 0, stream>>>(src, dst, ex2, feat2, agg2,
                                                     (long)E * DIM / 4, H2, F2);

  // ---- classifier head ----
  finalize2<<<g((long)N, 4), 128, 0, stream>>>(agg2, b2, fcW, fcb, thr, (float*)d_out, N);
}
